// FrameTransformer_8581344658019
// MI455X (gfx1250) — matrix operand order repaired and checked by reading, not compiled, not run on hardware
//
#include <hip/hip_runtime.h>
#include <hip/hip_bf16.h>
#include <stdint.h>

// FrameTransformer block for MI455X (gfx1250, wave32, WMMA).
// GEMM operands f16, accumulation + residual/LN/softmax stream fp32.
// Inner GEMM: 128x64x32 block tile, double-buffered LDS staged with
// global_load_async_to_lds_b128 (ASYNCcnt), 4x v_wmma_f32_16x16x32_f16
// per wave per k-step. Workspace need: ~283 MiB.

#ifndef USE_ASYNC
#define USE_ASYNC 1
#endif

typedef _Float16 h16;
typedef __attribute__((ext_vector_type(16))) _Float16 v16h;
typedef __attribute__((ext_vector_type(8)))  _Float16 v8h;
typedef __attribute__((ext_vector_type(4)))  _Float16 v4h;
typedef __attribute__((ext_vector_type(8)))  float    v8f;
typedef __attribute__((ext_vector_type(4)))  float    v4f;

#define EPSV 1e-5f

// ---------------------------------------------------------------------------
// Block reductions (256 threads = 8 waves, wave32)
// ---------------------------------------------------------------------------
__device__ __forceinline__ void blockReduce2(float& a, float& b) {
  __shared__ float sa[8], sb[8];
  __syncthreads();
  for (int o = 16; o > 0; o >>= 1) {
    a += __shfl_xor(a, o, 32);
    b += __shfl_xor(b, o, 32);
  }
  int lane = threadIdx.x & 31, wv = threadIdx.x >> 5;
  if (lane == 0) { sa[wv] = a; sb[wv] = b; }
  __syncthreads();
  if (threadIdx.x == 0) {
    float ta = 0.f, tb = 0.f;
    for (int i = 0; i < 8; i++) { ta += sa[i]; tb += sb[i]; }
    sa[0] = ta; sb[0] = tb;
  }
  __syncthreads();
  a = sa[0]; b = sb[0];
}

__device__ __forceinline__ float blockReduceMax(float v) {
  __shared__ float sm[8];
  __syncthreads();
  for (int o = 16; o > 0; o >>= 1) v = fmaxf(v, __shfl_xor(v, o, 32));
  int lane = threadIdx.x & 31, wv = threadIdx.x >> 5;
  if (lane == 0) sm[wv] = v;
  __syncthreads();
  if (threadIdx.x == 0) {
    float m = sm[0];
    for (int i = 1; i < 8; i++) m = fmaxf(m, sm[i]);
    sm[0] = m;
  }
  __syncthreads();
  return sm[0];
}

__device__ __forceinline__ float blockReduceSum(float v) {
  __shared__ float ss[8];
  __syncthreads();
  for (int o = 16; o > 0; o >>= 1) v += __shfl_xor(v, o, 32);
  int lane = threadIdx.x & 31, wv = threadIdx.x >> 5;
  if (lane == 0) ss[wv] = v;
  __syncthreads();
  if (threadIdx.x == 0) {
    float s = 0.f;
    for (int i = 0; i < 8; i++) s += ss[i];
    ss[0] = s;
  }
  __syncthreads();
  return ss[0];
}

// ---------------------------------------------------------------------------
// f32 -> f16 converter; n4 = elementCount/4
// ---------------------------------------------------------------------------
__global__ void k_cvt_f16(const float* __restrict__ s, h16* __restrict__ d,
                          long long n4) {
  long long i = (long long)blockIdx.x * 256 + threadIdx.x;
  if (i >= n4) return;
  v4f v = ((const v4f*)s)[i];
  v4h o;
  o[0] = (h16)v[0]; o[1] = (h16)v[1]; o[2] = (h16)v[2]; o[3] = (h16)v[3];
  ((v4h*)d)[i] = o;
}

// ---------------------------------------------------------------------------
// Generic batched WMMA GEMM:  C = act(alpha * A(MxK) * B^T(NxK))
//   A f16: element (m,k) at A + m*lda_m + k            (k contiguous)
//   B f16: element (n,k) at B + n*ldb_n + k*ldb_k
//   C: f32 (outHalf=0) or f16 (outHalf=1)
// Block tile 128(M)x64(N)x32(K), 8 waves; wave wv owns M rows [wv*16,+16)
// and all 64 N columns: 1 A frag x 4 B frags -> 4 WMMAs / wave / k-step.
// LDS double-buffered; tiles filled by global_load_async_to_lds_b128 and
// drained with s_wait_asynccnt (async loads complete in order).
// Requires M%128==0, N%64==0, K%32==0, 16-B aligned operand bases.
// ---------------------------------------------------------------------------
__global__ __launch_bounds__(256) void k_gemm_wmma(
    const h16* __restrict__ A, const h16* __restrict__ B,
    void* __restrict__ Cv,
    int M, int N, int K,
    long long lda_m, long long ldb_n, long long ldb_k, long long ldc,
    int bDiv,
    long long sA_b, long long sA_n,
    long long sB_b, long long sB_n,
    long long sC_b, long long sC_n,
    float alpha, int relu, int outHalf) {
  constexpr int LDT = 40;            // 80-B rows: every frag slice 16-B aligned
  constexpr int ASZ = 128 * LDT;     // one A buffer (halves)
  constexpr int BSZ = 64 * LDT;      // one B buffer (halves)
  __shared__ h16 As[2 * ASZ];        // As[buf][m][k]
  __shared__ h16 Bs[2 * BSZ];        // Bs[buf][n][k]

  const int z = blockIdx.z, bb = z / bDiv, nn = z % bDiv;
  const h16* Ap = A + bb * sA_b + nn * sA_n;
  const h16* Bp = B + bb * sB_b + nn * sB_n;

  const int m0 = blockIdx.y * 128, n0 = blockIdx.x * 64;
  const int tid = threadIdx.x, lane = tid & 31, wv = tid >> 5;

  // fragment coordinates (ISA 7.12.2)
  const int mloc = wv * 16 + (lane & 15);
  const int kbA  = (lane >> 4) * 8;
  const int nlocB = lane & 15;
  const int kbB  = (lane >> 4) * 8;

  // staging coordinates: one 16-B segment per thread (x2 for the A tile)
  const int bk = tid >> 3, bnseg = (tid & 7) << 3;    // act-B: 32 k x 8 segs

  auto stage = [&](int buf, int kk) {
    h16* as = As + buf * ASZ;
    h16* bs = Bs + buf * BSZ;
    // A tile: 128 rows x 32 k = 512 segments -> 2 per thread
    for (int s = 0; s < 2; s++) {
      int idx = tid + (s << 8);
      int row = idx >> 2, seg = (idx & 3) << 3;
      const h16* src = Ap + (long long)(m0 + row) * lda_m + (kk + seg);
#if USE_ASYNC
      asm volatile("global_load_async_to_lds_b128 %0, %1, off"
                   :: "v"((unsigned)(uintptr_t)&as[row * LDT + seg]),
                      "v"((unsigned long long)(uintptr_t)src)
                   : "memory");
#else
      *(v8h*)&as[row * LDT + seg] = *(const v8h*)src;
#endif
    }
    if (ldb_k == 1) {  // weight-style: k fast in memory -> async b128
      int row = tid >> 2, seg = (tid & 3) << 3;
      const h16* src = Bp + (long long)(n0 + row) * ldb_n + (kk + seg);
#if USE_ASYNC
      asm volatile("global_load_async_to_lds_b128 %0, %1, off"
                   :: "v"((unsigned)(uintptr_t)&bs[row * LDT + seg]),
                      "v"((unsigned long long)(uintptr_t)src)
                   : "memory");
#else
      *(v8h*)&bs[row * LDT + seg] = *(const v8h*)src;
#endif
    } else {           // activation-style: n fast -> v8h load + 8 LDS scatters
      v8h vb = *(const v8h*)(Bp + (long long)(kk + bk) * ldb_k + (n0 + bnseg));
      for (int e = 0; e < 8; e++) bs[(bnseg + e) * LDT + bk] = vb[e];
    }
  };

  v8f acc[4] = {};
  int cur = 0;
  stage(cur, 0);

  for (int k0 = 0; k0 < K; k0 += 32) {
    const bool more = (k0 + 32) < K;
    if (more) stage(cur ^ 1, k0 + 32);
#if USE_ASYNC
    // drain the current buffer's asyncs; allow the just-issued next-tile
    // asyncs (3 weight-mode / 2 act-mode) to stay outstanding.
    if (!more)          asm volatile("s_wait_asynccnt 0x0" ::: "memory");
    else if (ldb_k == 1) asm volatile("s_wait_asynccnt 0x3" ::: "memory");
    else                 asm volatile("s_wait_asynccnt 0x2" ::: "memory");
#endif
    __syncthreads();

    const h16* as = As + cur * ASZ;
    const h16* bs = Bs + cur * BSZ;
    v16h af;
    {
      v8h lo = *(const v8h*)&as[mloc * LDT + kbA];
      v8h hi = *(const v8h*)&as[mloc * LDT + kbA + 16];
      for (int e = 0; e < 8; e++) { af[e] = lo[e]; af[e + 8] = hi[e]; }
    }
    for (int j = 0; j < 4; j++) {
      const h16* bp = &bs[(j * 16 + nlocB) * LDT + kbB];
      v16h bf;
      v8h lo = *(const v8h*)bp;
      v8h hi = *(const v8h*)(bp + 16);
      for (int e = 0; e < 8; e++) { bf[e] = lo[e]; bf[e + 8] = hi[e]; }
      acc[j] = __builtin_amdgcn_wmma_f32_16x16x32_f16(
          false, af, false, bf, (short)0, acc[j], false, false);
    }
    __syncthreads();   // everyone done reading `cur` before it is re-staged
    cur ^= 1;
  }

  // ---- epilogue (C 16x16 f32 layout: VGPR r -> M=r / r+8, N=lane%16) ----
  const int mrow = m0 + wv * 16 + (lane >> 4) * 8;
  const int ncb  = n0 + (lane & 15);
  if (outHalf) {
    h16* C = (h16*)Cv + bb * sC_b + nn * sC_n;
    for (int j = 0; j < 4; j++)
      for (int r = 0; r < 8; r++) {
        float v = alpha * acc[j][r];
        if (relu) v = fmaxf(v, 0.f);
        C[(long long)(mrow + r) * ldc + ncb + j * 16] = (h16)v;
      }
  } else {
    float* C = (float*)Cv + bb * sC_b + nn * sC_n;
    for (int j = 0; j < 4; j++)
      for (int r = 0; r < 8; r++) {
        float v = alpha * acc[j][r];
        if (relu) v = fmaxf(v, 0.f);
        C[(long long)(mrow + r) * ldc + ncb + j * 16] = v;
      }
  }
}

// ---------------------------------------------------------------------------
// Bottleneck: xs[b,w,bin] = relu(sum_c x[b,c,bin,w]*bw[c] / sqrt(1+eps))
// 32x32 LDS transpose; writes fp32 XS and f16 XSh.
// ---------------------------------------------------------------------------
__global__ void k_bottleneck(const float* __restrict__ x,
                             const float* __restrict__ bw,
                             float* __restrict__ XS, h16* __restrict__ XSh) {
  __shared__ float tile[32][33];
  __shared__ float wsm[8];
  const int b = blockIdx.z, bin0 = blockIdx.y * 32, w0 = blockIdx.x * 32;
  const int tx = threadIdx.x & 31, ty = threadIdx.x >> 5;
  if (threadIdx.x < 8) wsm[threadIdx.x] = bw[threadIdx.x];
  __syncthreads();
  const float scale = rsqrtf(1.0f + EPSV);
  for (int r = 0; r < 4; r++) {
    int bin = bin0 + ty + r * 8;
    float acc = 0.f;
    for (int c = 0; c < 8; c++)
      acc += x[(((long long)(b * 8 + c) << 10) + bin) * 1024 + w0 + tx] * wsm[c];
    tile[ty + r * 8][tx] = fmaxf(acc * scale, 0.f);
  }
  __syncthreads();
  for (int r = 0; r < 4; r++) {
    int w = w0 + ty + r * 8;
    long long o = (((long long)b << 10) + w) * 1024 + bin0 + tx;
    float v = tile[tx][ty + r * 8];
    XS[o] = v;
    XSh[o] = (h16)v;
  }
}

// ---------------------------------------------------------------------------
// GLU + residual + LN: xs = ln(xs + g[:1024]*sigmoid(g[1024:])); also f16 copy
// ---------------------------------------------------------------------------
__global__ void k_glu_res_ln(float* __restrict__ XS, const float* __restrict__ G,
                             h16* __restrict__ XSh) {
  const long long row = blockIdx.x;
  const float* g  = G  + row * 2048;
  float*       xs = XS + row * 1024;
  h16*         xh = XSh + row * 1024;
  float t[4], s = 0.f, sq = 0.f;
  for (int i = 0; i < 4; i++) {
    int c = threadIdx.x + (i << 8);
    float gate = g[c + 1024];
    float v = xs[c] + g[c] * (1.f / (1.f + __expf(-gate)));
    t[i] = v; s += v; sq += v * v;
  }
  blockReduce2(s, sq);
  float mean = s * (1.f / 1024.f);
  float inv  = rsqrtf(sq * (1.f / 1024.f) - mean * mean + EPSV);
  for (int i = 0; i < 4; i++) {
    int c = threadIdx.x + (i << 8);
    float v = (t[i] - mean) * inv;
    xs[c] = v; xh[c] = (h16)v;
  }
}

// ---------------------------------------------------------------------------
// ln(x + pad(h)): optional fp32 in-place write (writeX), optional f16 out (Xh)
// ---------------------------------------------------------------------------
__global__ void k_add_ln(float* __restrict__ X, const float* __restrict__ H,
                         int W, int addW, int writeX, h16* __restrict__ Xh) {
  const long long row = blockIdx.x;
  float*       xr = X + row * (long long)W;
  const float* hr = H + row * (long long)addW;
  const int EL = W >> 8;
  float t[16], s = 0.f, sq = 0.f;
  for (int i = 0; i < EL; i++) {
    int c = threadIdx.x + (i << 8);
    float v = xr[c] + (c < addW ? hr[c] : 0.f);
    t[i] = v; s += v; sq += v * v;
  }
  blockReduce2(s, sq);
  float rw = 1.f / (float)W;
  float mean = s * rw;
  float inv  = rsqrtf(sq * rw - mean * mean + EPSV);
  h16* xh = Xh ? Xh + row * (long long)W : nullptr;
  for (int i = 0; i < EL; i++) {
    int c = threadIdx.x + (i << 8);
    float v = (t[i] - mean) * inv;
    if (writeX) xr[c] = v;
    if (xh) xh[c] = (h16)v;
  }
}

// ---------------------------------------------------------------------------
// im2col (K=3, pad=1 over w), f16 in / f16 out: T[bw, c*3+t] = xs[b,w+t-1,c]
// ---------------------------------------------------------------------------
__global__ void k_im2col(const h16* __restrict__ XSh, h16* __restrict__ T) {
  long long idx = (long long)blockIdx.x * 256 + threadIdx.x;
  int col = (int)(idx % 3072);
  long long row = idx / 3072;
  int b = (int)(row >> 10), w = (int)(row & 1023);
  int t = col % 3, c = col / 3;
  int ws = w + t - 1;
  h16 v = (h16)0.f;
  if (ws >= 0 && ws < 1024)
    v = XSh[(((long long)(b << 10) + ws) << 10) + c];
  T[idx] = v;
}

// ---------------------------------------------------------------------------
// depthwise conv k=9 pad=4 over w (layout [b,w,c]), f16 in/out, fp32 math
// ---------------------------------------------------------------------------
__global__ void k_dwconv9(const h16* __restrict__ Hin,
                          const float* __restrict__ W,
                          h16* __restrict__ Hout) {
  long long idx = (long long)blockIdx.x * 256 + threadIdx.x;
  int c = (int)(idx & 4095);
  long long row = idx >> 12;
  int b = (int)(row >> 10), w = (int)(row & 1023);
  float acc = 0.f;
  for (int t = 0; t < 9; t++) {
    int ws = w + t - 4;
    if (ws >= 0 && ws < 1024)
      acc += (float)Hin[(((long long)(b << 10) + ws) << 12) + c] * W[c * 9 + t];
  }
  Hout[idx] = (h16)acc;
}

// ---------------------------------------------------------------------------
// softmax(s + |i-j|/2*dw[n,i] + qe^T) -> f16 probs
// (pad/swap/slice skew trick == transpose: skew[i,j] = qe[b,n,j,i])
// ---------------------------------------------------------------------------
__global__ void k_softmax_bias(const float* __restrict__ S,
                               const float* __restrict__ QE,
                               const float* __restrict__ DW,
                               h16* __restrict__ P) {
  const int row = blockIdx.x;          // ((b*4+n)<<10) + i
  const int i  = row & 1023;
  const int bn = row >> 10;
  const int n  = bn & 3;
  const float dwv = DW[(n << 10) + i];
  const float* s = S + (long long)row * 1024;
  const float* qe = QE + ((long long)bn << 20) + i;  // + j*1024
  h16* p = P + (long long)row * 1024;
  float t[4], mx = -1e30f;
  for (int k = 0; k < 4; k++) {
    int j = threadIdx.x + (k << 8);
    float v = s[j] + fabsf((float)(i - j)) * 0.5f * dwv + qe[(long long)j << 10];
    t[k] = v; mx = fmaxf(mx, v);
  }
  mx = blockReduceMax(mx);
  float sum = 0.f;
  for (int k = 0; k < 4; k++) { t[k] = __expf(t[k] - mx); sum += t[k]; }
  sum = blockReduceSum(sum);
  float inv = 1.f / sum;
  for (int k = 0; k < 4; k++) {
    int j = threadIdx.x + (k << 8);
    p[j] = (h16)(t[k] * inv);
  }
}

// ---------------------------------------------------------------------------
// out[b,0,bin,w] = xs[b,w,bin]
// ---------------------------------------------------------------------------
__global__ void k_transpose_out(const float* __restrict__ XS,
                                float* __restrict__ out) {
  __shared__ float tile[32][33];
  const int b = blockIdx.z, w0 = blockIdx.x * 32, bin0 = blockIdx.y * 32;
  const int tx = threadIdx.x & 31, ty = threadIdx.x >> 5;
  for (int r = 0; r < 4; r++)
    tile[ty + r * 8][tx] =
        XS[(((long long)(b << 10) + w0 + ty + r * 8) << 10) + bin0 + tx];
  __syncthreads();
  for (int r = 0; r < 4; r++)
    out[(((long long)(b << 10) + bin0 + ty + r * 8) << 10) + w0 + tx] =
        tile[tx][ty + r * 8];
}

// ---------------------------------------------------------------------------
extern "C" void kernel_launch(void* const* d_in, const int* in_sizes, int n_in,
                              void* d_out, int out_size, void* d_ws, size_t ws_size,
                              hipStream_t stream) {
  const float* x        = (const float*)d_in[0];
  const float* bneck_w  = (const float*)d_in[1];
  const float* glu_w    = (const float*)d_in[2];
  const float* conv1L_w = (const float*)d_in[3];
  const float* conv1R_w = (const float*)d_in[4];
  const float* conv2_dw = (const float*)d_in[5];
  const float* conv2_pw = (const float*)d_in[6];
  const float* q_w      = (const float*)d_in[7];
  const float* k_w      = (const float*)d_in[8];
  const float* v_w      = (const float*)d_in[9];
  const float* o_w      = (const float*)d_in[10];
  const float* er       = (const float*)d_in[11];
  const float* dist_w   = (const float*)d_in[12];
  const float* conv3_w  = (const float*)d_in[13];
  const float* conv4_w  = (const float*)d_in[14];
  float* out = (float*)d_out;

  // ---- workspace carve-up (256-B aligned bump allocator) ----
  char* base = (char*)d_ws;
  size_t off = 0;
  auto alloc = [&](size_t bytes) {
    char* p = base + off;
    off = (off + bytes + 255) & ~(size_t)255;
    return p;
  };
  const long long M1 = 1ll << 20;                    // 1024*1024
  float* XS   = (float*)alloc(4 * M1 * 4);           // fp32 stream
  h16*   XSh  = (h16*)  alloc(4 * M1 * 2);           // f16 mirror
  float* T1   = (float*)alloc(16 * M1 * 4);          // GLU out / hL / scores
  float* T3   = (float*)alloc(4 * M1 * 4);           // small fp32 results
  char*  RA   = alloc(16 * M1 * 4);                  // H1h+DWH, later QE
  h16*   H1h  = (h16*)RA;                            // [4096,4096] f16
  h16*   DWH  = (h16*)(RA + 16 * M1 * 2);            // [4096,4096] f16
  float* QE   = (float*)RA;                          // [4,4,1024,1024] fp32
  char*  RB   = alloc(16 * M1 * 2);                  // IM2h / Ph / H3h
  h16*   IM2h = (h16*)RB;                            // [4096,3072]
  h16*   Ph   = (h16*)RB;                            // [16,1024,1024]
  h16*   H3h  = (h16*)RB;                            // [4096,4096]
  h16*   Qh   = (h16*)alloc(4 * M1 * 2);
  h16*   Kh   = (h16*)alloc(4 * M1 * 2);
  h16*   Vh   = (h16*)alloc(4 * M1 * 2);
  h16*   CTXh = (h16*)alloc(4 * M1 * 2);
  h16* glu_wh = (h16*)alloc(2 * M1 * 2);
  h16* c1Lh   = (h16*)alloc(4 * M1 * 2);
  h16* c1Rh   = (h16*)alloc(3 * M1 * 2);
  h16* c2pwh  = (h16*)alloc(4 * M1 * 2);
  h16* qwh    = (h16*)alloc(M1 * 2);
  h16* kwh    = (h16*)alloc(M1 * 2);
  h16* vwh    = (h16*)alloc(M1 * 2);
  h16* owh    = (h16*)alloc(M1 * 2);
  h16* erh    = (h16*)alloc((M1 / 4) * 2);
  h16* c3h    = (h16*)alloc(4 * M1 * 2);
  h16* c4h    = (h16*)alloc(4 * M1 * 2);

  auto cvt = [&](const float* s, h16* d, long long n) {
    k_cvt_f16<<<(int)(n / 4 / 256), 256, 0, stream>>>(s, d, n / 4);
  };
  auto gemm = [&](const h16* A, const h16* B, void* C,
                  int M, int N, int K,
                  long long lda_m, long long ldb_n, long long ldb_k,
                  long long ldc, float alpha, int relu, int outHalf,
                  int batch = 1, int bDiv = 1,
                  long long sA_b = 0, long long sA_n = 0,
                  long long sB_b = 0, long long sB_n = 0,
                  long long sC_b = 0, long long sC_n = 0) {
    dim3 grid(N / 64, M / 128, batch);
    k_gemm_wmma<<<grid, 256, 0, stream>>>(A, B, C, M, N, K, lda_m, ldb_n,
                                          ldb_k, ldc, bDiv, sA_b, sA_n, sB_b,
                                          sB_n, sC_b, sC_n, alpha, relu,
                                          outHalf);
  };

  // ---- weight conversions (once per call) ----
  cvt(glu_w,    glu_wh, 2 * M1);
  cvt(conv1L_w, c1Lh,   4 * M1);
  cvt(conv1R_w, c1Rh,   3 * M1);
  cvt(conv2_pw, c2pwh,  4 * M1);
  cvt(q_w, qwh, M1); cvt(k_w, kwh, M1); cvt(v_w, vwh, M1); cvt(o_w, owh, M1);
  cvt(er, erh, M1 / 4);
  cvt(conv3_w, c3h, 4 * M1);
  cvt(conv4_w, c4h, 4 * M1);

  // 0. bottleneck + transpose -> XS / XSh [b,w,bins]
  k_bottleneck<<<dim3(32, 32, 4), 256, 0, stream>>>(x, bneck_w, XS, XSh);
  // 1. GLU projection, epilogue + ln
  gemm(XSh, glu_wh, T1, 4096, 2048, 1024, 1024, 1024, 1, 2048, 1.f, 0, 0);
  k_glu_res_ln<<<4096, 256, 0, stream>>>(XS, T1, XSh);
  // 2. hL = relu(xs @ conv1L.T) -> T1 fp32
  gemm(XSh, c1Lh, T1, 4096, 4096, 1024, 1024, 1024, 1, 4096, 1.f, 1, 0);
  // 3. hR = relu(conv1d(xs, K=3)) via f16 im2col GEMM (K=3072)
  k_im2col<<<49152, 256, 0, stream>>>(XSh, IM2h);
  gemm(IM2h, c1Rh, T3, 4096, 1024, 3072, 3072, 3072, 1, 1024, 1.f, 1, 0);
  // 4. h = ln(hL + pad(hR)) -> f16 only (feeds depthwise conv)
  k_add_ln<<<4096, 256, 0, stream>>>(T1, T3, 4096, 1024, 0, H1h);
  // 5. depthwise k=9 conv over w
  k_dwconv9<<<65536, 256, 0, stream>>>(H1h, conv2_dw, DWH);
  // 6. pointwise 4096 -> 1024
  gemm(DWH, c2pwh, T3, 4096, 1024, 4096, 4096, 4096, 1, 1024, 1.f, 0, 0);
  // 7. xs = ln(xs + h)
  k_add_ln<<<4096, 256, 0, stream>>>(XS, T3, 1024, 1024, 1, XSh);
  // 8. Q, K, V projections (f16 outputs)
  gemm(XSh, qwh, Qh, 4096, 1024, 1024, 1024, 1024, 1, 1024, 1.f, 0, 1);
  gemm(XSh, kwh, Kh, 4096, 1024, 1024, 1024, 1024, 1, 1024, 1.f, 0, 1);
  gemm(XSh, vwh, Vh, 4096, 1024, 1024, 1024, 1024, 1, 1024, 1.f, 0, 1);
  // 9. qe[b,n] = Q_band @ er  (er[d,cs]: ldb_n=1, ldb_k=1024) -> fp32
  gemm(Qh, erh, QE, 1024, 1024, 256, 1024, 1, 1024, 1024, 1.f, 0, 0,
       16, 4, M1, 256, 0, 0, 4 * M1, M1);
  // 10. scores = Q_band @ K_band^T / 32 -> fp32 T1
  gemm(Qh, Kh, T1, 1024, 1024, 256, 1024, 1024, 1, 1024, 1.f / 32.f, 0, 0,
       16, 4, M1, 256, M1, 256, 4 * M1, M1);
  // 11. softmax(scores + dist*dw + qe^T) -> f16 probs
  k_softmax_bias<<<16384, 256, 0, stream>>>(T1, QE, dist_w, Ph);
  // 12. context = P @ V_band (v[j,d]: ldb_n=1, ldb_k=1024) -> f16
  gemm(Ph, Vh, CTXh, 1024, 256, 1024, 1024, 1, 1024, 1024, 1.f, 0, 1,
       16, 4, 4 * M1, M1, M1, 256, M1, 256);
  // 13. out projection, residual + ln
  gemm(CTXh, owh, T3, 4096, 1024, 1024, 1024, 1024, 1, 1024, 1.f, 0, 0);
  k_add_ln<<<4096, 256, 0, stream>>>(XS, T3, 1024, 1024, 1, XSh);
  // 14. final FFN
  gemm(XSh, c3h, H3h, 4096, 4096, 1024, 1024, 1024, 1, 4096, 1.f, 1, 1);
  gemm(H3h, c4h, T3, 4096, 1024, 4096, 4096, 4096, 1, 1024, 1.f, 0, 0);
  k_add_ln<<<4096, 256, 0, stream>>>(XS, T3, 1024, 1024, 1, nullptr);
  // 15. transpose to [b,1,bins,w]
  k_transpose_out<<<dim3(32, 32, 4), 256, 0, stream>>>(XS, out);
}
